// MPNN_53970559042214
// MI455X (gfx1250) — matrix operand order repaired and checked by reading, not compiled, not run on hardware
//
#include <hip/hip_runtime.h>

// ---------------------------------------------------------------------------
// CDNA5 (gfx1250) types for WMMA: wave32, V_WMMA_F32_16X16X32_BF16
// ---------------------------------------------------------------------------
typedef __bf16          v16bf __attribute__((ext_vector_type(16)));
typedef float           v8f   __attribute__((ext_vector_type(8)));
typedef unsigned int    v8u   __attribute__((ext_vector_type(8)));
typedef unsigned short  v16us __attribute__((ext_vector_type(16)));

__device__ __forceinline__ unsigned short f2bf(float f) {
  union { float f; unsigned u; } v; v.f = f;
  unsigned u = v.u;
  unsigned r = u + 0x7fffu + ((u >> 16) & 1u);   // round-to-nearest-even
  return (unsigned short)(r >> 16);
}
__device__ __forceinline__ float bf2f(unsigned short b) {
  union { unsigned u; float f; } v; v.u = ((unsigned)b) << 16;
  return v.f;
}
__device__ __forceinline__ float sigmoidf_(float x) { return 1.0f / (1.0f + expf(-x)); }

// ordered-uint encoding of float for atomicMax-based segment max
__device__ __forceinline__ unsigned encf(float f) {
  union { float f; unsigned u; } v; v.f = f;
  return (v.u & 0x80000000u) ? ~v.u : (v.u | 0x80000000u);
}
__device__ __forceinline__ float decf(unsigned e) {
  union { unsigned u; float f; } v;
  v.u = (e & 0x80000000u) ? (e & 0x7fffffffu) : ~e;
  return v.f;
}

// packed 2xbf16 multiply, native RNE (V_PK_MUL_BF16, VOP3P op 42)
__device__ __forceinline__ unsigned pk_mul_bf16(unsigned a, unsigned b) {
  unsigned d;
  asm("v_pk_mul_bf16 %0, %1, %2" : "=v"(d) : "v"(a), "v"(b));
  return d;
}

// ---------------------------------------------------------------------------
// Generic bf16 WMMA GEMM: C[M,N] = act(A[M,Kp] @ B[Kp,N] + bias)
// A row-major (lda), B row-major (ldb=N), Kp multiple of 32, N = NT*16.
// One wave per 16-row tile; 4 waves per block.
// A fragment (ISA 7.12.2, 16-bit A 16x32): lane&15 = M row, half-waves split K.
// B fragment: lane&15 = N column, element s = K row (same K order as A).
// C fragment: VGPR r -> row r + 8*(lane>=16), col = lane&15.
// ---------------------------------------------------------------------------
template <int NT>
__global__ void __launch_bounds__(128)
gemm_bf16_wmma(const unsigned short* __restrict__ A,
               const unsigned short* __restrict__ Bm,
               const float* __restrict__ bias,
               float* __restrict__ Cf,
               unsigned short* __restrict__ Cb,
               int M, int Kp, int lda, int act)
{
  constexpr int N = NT * 16;
  const int lane   = threadIdx.x & 31;
  const int wave   = threadIdx.x >> 5;
  const int mtile  = blockIdx.x * 4 + wave;
  const int mtiles = (M + 15) >> 4;
  if (mtile >= mtiles) return;          // wave-uniform: EXEC stays all-ones
  const int khalf = lane >> 4;
  const int row   = (mtile << 4) + (lane & 15);

  v8f acc[NT];
#pragma unroll
  for (int nt = 0; nt < NT; ++nt)
#pragma unroll
    for (int i = 0; i < 8; ++i) acc[nt][i] = 0.0f;

  const int nkc = Kp >> 5;
  for (int kc = 0; kc < nkc; ++kc) {
    v8u au;
    if (row < M) {
      const unsigned short* ap = A + (size_t)row * lda + (kc << 5) + (khalf << 3);
#pragma unroll
      for (int v = 0; v < 8; ++v) {
        const int K = ((v & 3) << 1) + ((v >> 2) << 4);  // {0,2,4,6,16,18,20,22}
        au[v] = *(const unsigned int*)(ap + K);
      }
    } else {
#pragma unroll
      for (int v = 0; v < 8; ++v) au[v] = 0u;
    }
    v16bf afrag = __builtin_bit_cast(v16bf, au);
    const unsigned short* bp = Bm + (size_t)(kc << 5) * N + (lane & 15);
#pragma unroll
    for (int nt = 0; nt < NT; ++nt) {
      v16us braw;
#pragma unroll
      for (int s = 0; s < 16; ++s)
        braw[s] = bp[(size_t)((khalf << 3) + (s & 7) + ((s >> 3) << 4)) * N + nt * 16];
      v16bf bfrag = __builtin_bit_cast(v16bf, braw);
      acc[nt] = __builtin_amdgcn_wmma_f32_16x16x32_bf16(
          false, afrag, false, bfrag, (short)0, acc[nt], false, false);
    }
  }

#pragma unroll
  for (int nt = 0; nt < NT; ++nt) {
    const int col = nt * 16 + (lane & 15);
    const float bv = bias ? bias[col] : 0.0f;
#pragma unroll
    for (int r = 0; r < 8; ++r) {
      const int orow = (mtile << 4) + r + (khalf << 3);
      if (orow < M) {
        float val = acc[nt][r] + bv;
        if (act) val = fmaxf(val, 0.0f);
        if (Cf) Cf[(size_t)orow * N + col] = val;
        if (Cb) Cb[(size_t)orow * N + col] = f2bf(val);
      }
    }
  }
}

// ---------------------------------------------------------------------------
// Fused NNConv message GEMM:
//   msg[e,o] = sum_{k,i} hid[e,k]*xsrc[e,i]*W2r[k*64+i,o]  (+ bias rows 8192..8255)
// The implicit P = h (outer) x row is generated in registers: per 32-wide K
// chunk the needed x values are two contiguous 16B LDS reads, multiplied by a
// lane-splatted h[k] with native packed v_pk_mul_bf16 (8 ops per fragment).
// 258 K-chunks of 32 (256 product chunks + 2 bias chunks), 4 N-tiles.
// Result scattered with atomicAdd into agg[dst[e], :].  W2r (1 MB) stays
// L2-resident and is reused by all 3750 edge tiles -> compute bound.
// ---------------------------------------------------------------------------
__global__ void __launch_bounds__(128)
k_msg(const unsigned short* __restrict__ hid,    // [E,128] bf16
      const unsigned short* __restrict__ xsrc,   // [E,64]  bf16
      const unsigned short* __restrict__ W2r,    // [8256,64] bf16 (L2-resident)
      const int* __restrict__ dst,
      float* __restrict__ agg,                   // [N,64] f32
      int E_)
{
  __shared__ unsigned short xs[4][16 * 64];
  __shared__ unsigned short hs[4][16 * 128];
  const int lane   = threadIdx.x & 31;
  const int wave   = threadIdx.x >> 5;
  const int tile   = blockIdx.x * 4 + wave;
  const int ntiles = E_ >> 4;
  const bool active = tile < ntiles;
  const int e0 = tile << 4;

  if (active) {
    const unsigned* gx = (const unsigned*)(xsrc + (size_t)e0 * 64);
    unsigned* lx = (unsigned*)xs[wave];
#pragma unroll
    for (int t = 0; t < 16; ++t) lx[t * 32 + lane] = gx[t * 32 + lane];
    const unsigned* gh2 = (const unsigned*)(hid + (size_t)e0 * 128);
    unsigned* lh = (unsigned*)hs[wave];
#pragma unroll
    for (int t = 0; t < 32; ++t) lh[t * 32 + lane] = gh2[t * 32 + lane];
  }
  __syncthreads();
  if (!active) return;

  const int rrow  = lane & 15;
  const int khalf = lane >> 4;
  const unsigned short* xrow = xs[wave] + rrow * 64;
  const unsigned short* hrow = hs[wave] + rrow * 128;

  v8f acc[4];
#pragma unroll
  for (int nt = 0; nt < 4; ++nt)
#pragma unroll
    for (int i = 0; i < 8; ++i) acc[nt][i] = 0.0f;

  for (int kc = 0; kc < 258; ++kc) {
    // x values needed by this lane's A fragment: two contiguous 16B groups
    const int ibase = (kc < 256) ? ((kc & 1) << 5) : ((kc - 256) << 5);
    const unsigned short* xp0 = xrow + ibase + (khalf << 3);
    v8u xv;
#pragma unroll
    for (int v = 0; v < 8; ++v)
      xv[v] = *(const unsigned int*)(xp0 + ((v >> 2) << 4) + ((v & 3) << 1));

    v8u au;
    if (kc < 256) {
      const unsigned hbits = (unsigned)hrow[kc >> 1];    // k index = kc/2
      const unsigned hsp   = (hbits << 16) | hbits;      // splat to both halves
#pragma unroll
      for (int v = 0; v < 8; ++v)
        au[v] = pk_mul_bf16(xv[v], hsp);                 // 2 bf16 products / op
    } else {
      au = xv;                                           // bias rows: P = xsrc
    }
    v16bf afrag = __builtin_bit_cast(v16bf, au);

    const unsigned short* bp = W2r + (size_t)(kc * 32) * 64 + rrow;
    if (kc + 1 < 258) __builtin_prefetch(bp + 32 * 64, 0, 1);
#pragma unroll
    for (int nt = 0; nt < 4; ++nt) {
      v16us braw;
#pragma unroll
      for (int s = 0; s < 16; ++s)
        braw[s] = bp[((khalf << 3) + (s & 7) + ((s >> 3) << 4)) * 64 + nt * 16];
      v16bf bfrag = __builtin_bit_cast(v16bf, braw);
      acc[nt] = __builtin_amdgcn_wmma_f32_16x16x32_bf16(
          false, afrag, false, bfrag, (short)0, acc[nt], false, false);
    }
  }

#pragma unroll
  for (int nt = 0; nt < 4; ++nt) {
    const int col = nt * 16 + rrow;
#pragma unroll
    for (int r = 0; r < 8; ++r) {
      const int e = e0 + r + (khalf << 3);
      const int dnode = dst[e];
      atomicAdd(&agg[(size_t)dnode * 64 + col], acc[nt][r]);
    }
  }
}

// ---------------------------------------------------------------------------
// Small elementwise / prep kernels
// ---------------------------------------------------------------------------
__global__ void k_cvt_pad(const float* __restrict__ src, unsigned short* __restrict__ dstp,
                          int srows, int scols, int drows, int dcols)
{
  int idx = blockIdx.x * blockDim.x + threadIdx.x;
  if (idx >= drows * dcols) return;
  int r = idx / dcols, c = idx - r * dcols;
  float v = (r < srows && c < scols) ? src[r * scols + c] : 0.0f;
  dstp[idx] = f2bf(v);
}

// W2r[l][k*64+i][o] = enn2_W[l][k][i*64+o]; rows 8192..8255 = enn2_b reshaped
__global__ void k_build_w2r(const float* __restrict__ w2, const float* __restrict__ b2,
                            unsigned short* __restrict__ out)
{
  int idx = blockIdx.x * blockDim.x + threadIdx.x;
  const int total = 3 * 8256 * 64;
  if (idx >= total) return;
  int l = idx / (8256 * 64);
  int rem = idx - l * 8256 * 64;
  int row = rem >> 6;
  int o = rem & 63;
  float v;
  if (row < 8192) {
    int k = row >> 6, i = row & 63;
    v = w2[(size_t)l * 128 * 4096 + (size_t)k * 4096 + i * 64 + o];
  } else {
    int i = row - 8192;
    v = b2[l * 4096 + i * 64 + o];
  }
  out[idx] = f2bf(v);
}

__global__ void k_zero(float* __restrict__ p, int n) {
  int i = blockIdx.x * blockDim.x + threadIdx.x;
  if (i < n) p[i] = 0.0f;
}

__global__ void k_deg(const int* __restrict__ dst, float* __restrict__ deg, int E_) {
  int e = blockIdx.x * blockDim.x + threadIdx.x;
  if (e < E_) atomicAdd(&deg[dst[e]], 1.0f);
}

__global__ void k_gather(const unsigned short* __restrict__ hb, const int* __restrict__ src,
                         unsigned short* __restrict__ out, int total)
{
  int idx = blockIdx.x * blockDim.x + threadIdx.x;
  if (idx >= total) return;
  int e = idx >> 6, i = idx & 63;
  out[idx] = hb[(size_t)src[e] * 64 + i];
}

__global__ void k_m(const float* __restrict__ agg, const float* __restrict__ deg,
                    const float* __restrict__ troot, const float* __restrict__ convb,
                    unsigned short* __restrict__ mb, int total)
{
  int idx = blockIdx.x * blockDim.x + threadIdx.x;
  if (idx >= total) return;
  int n = idx >> 6, c = idx & 63;
  float d = fmaxf(deg[n], 1.0f);
  float v = agg[idx] / d + troot[idx] + convb[c];
  mb[idx] = f2bf(fmaxf(v, 0.0f));
}

__global__ void k_gru(const float* __restrict__ gi, const float* __restrict__ gh,
                      float* __restrict__ h, unsigned short* __restrict__ hb, int total)
{
  int idx = blockIdx.x * blockDim.x + threadIdx.x;
  if (idx >= total) return;
  int n = idx >> 6, c = idx & 63;
  const float* gir = gi + (size_t)n * 192;
  const float* ghr = gh + (size_t)n * 192;
  float r  = sigmoidf_(gir[c] + ghr[c]);
  float z  = sigmoidf_(gir[64 + c] + ghr[64 + c]);
  float nn = tanhf(gir[128 + c] + r * ghr[128 + c]);
  float hnew = (1.0f - z) * nn + z * h[idx];
  h[idx]  = hnew;
  hb[idx] = f2bf(hnew);
}

// naive f32 dense for the tiny head / LSTM gates: Y = [accum+] X@W + bias
__global__ void k_dense(const float* __restrict__ X, const float* __restrict__ W,
                        const float* __restrict__ bias, float* __restrict__ Y,
                        int Bn, int Kin, int Nout, int relu, int accum)
{
  int idx = blockIdx.x * blockDim.x + threadIdx.x;
  if (idx >= Bn * Nout) return;
  int b = idx / Nout, c = idx - b * Nout;
  float s = bias ? bias[c] : 0.0f;
  const float* xr = X + (size_t)b * Kin;
  for (int j = 0; j < Kin; ++j) s += xr[j] * W[(size_t)j * Nout + c];
  if (accum) s += Y[idx];
  if (relu) s = fmaxf(s, 0.0f);
  Y[idx] = s;
}

__global__ void k_lstm_update(const float* __restrict__ gates, float* __restrict__ hh,
                              float* __restrict__ cc, int total)
{
  int idx = blockIdx.x * blockDim.x + threadIdx.x;
  if (idx >= total) return;
  int b = idx >> 6, c = idx & 63;
  const float* g = gates + ((size_t)b << 8);
  float gi_ = g[c], gf_ = g[64 + c], gg_ = g[128 + c], go_ = g[192 + c];
  float cnew = sigmoidf_(gf_) * cc[idx] + sigmoidf_(gi_) * tanhf(gg_);
  cc[idx] = cnew;
  hh[idx] = sigmoidf_(go_) * tanhf(cnew);
}

__global__ void k_attn1(const float* __restrict__ h, const float* __restrict__ hh,
                        const int* __restrict__ batch, float* __restrict__ e_buf,
                        unsigned* __restrict__ emax, int n)
{
  int i = blockIdx.x * blockDim.x + threadIdx.x;
  if (i >= n) return;
  int b = batch[i];
  const float* hr = h + (size_t)i * 64;
  const float* qr = hh + (size_t)b * 64;
  float s = 0.0f;
  for (int d = 0; d < 64; ++d) s += hr[d] * qr[d];
  e_buf[i] = s;
  atomicMax(&emax[b], encf(s));
}

__global__ void k_attn2(float* __restrict__ e_buf, const unsigned* __restrict__ emax,
                        const int* __restrict__ batch, float* __restrict__ denom, int n)
{
  int i = blockIdx.x * blockDim.x + threadIdx.x;
  if (i >= n) return;
  int b = batch[i];
  float ee = expf(e_buf[i] - decf(emax[b]));
  e_buf[i] = ee;
  atomicAdd(&denom[b], ee);
}

__global__ void k_attn3(const float* __restrict__ e_buf, const float* __restrict__ denom,
                        const int* __restrict__ batch, const float* __restrict__ h,
                        float* __restrict__ rvec, int total)
{
  int idx = blockIdx.x * blockDim.x + threadIdx.x;
  if (idx >= total) return;
  int n = idx >> 6, d = idx & 63;
  int b = batch[n];
  float a = e_buf[n] / fmaxf(denom[b], 1e-16f);
  atomicAdd(&rvec[(size_t)b * 64 + d], a * h[idx]);
}

__global__ void k_qstar(const float* __restrict__ hh, const float* __restrict__ rvec,
                        float* __restrict__ qstar, int total)
{
  int idx = blockIdx.x * blockDim.x + threadIdx.x;
  if (idx >= total) return;
  int b = idx >> 7, c = idx & 127;
  qstar[idx] = (c < 64) ? hh[(size_t)b * 64 + c] : rvec[(size_t)b * 64 + c - 64];
}

// ---------------------------------------------------------------------------
// Orchestration
// ---------------------------------------------------------------------------
extern "C" void kernel_launch(void* const* d_in, const int* in_sizes, int n_in,
                              void* d_out, int out_size, void* d_ws, size_t ws_size,
                              hipStream_t stream)
{
  (void)in_sizes; (void)n_in; (void)out_size; (void)ws_size;
  constexpr int N = 15000, E = 60000;
  constexpr int Bg = 100;

  const float* x         = (const float*)d_in[0];
  const float* edge_attr = (const float*)d_in[1];
  const int*   eidx      = (const int*)d_in[2];
  const int*   batch     = (const int*)d_in[3];
  const float* lin0_W    = (const float*)d_in[4];
  const float* lin0_b    = (const float*)d_in[5];
  const float* enn1_W    = (const float*)d_in[6];
  const float* enn1_b    = (const float*)d_in[7];
  const float* enn2_W    = (const float*)d_in[8];
  const float* enn2_b    = (const float*)d_in[9];
  const float* root_W    = (const float*)d_in[10];
  const float* conv_b    = (const float*)d_in[11];
  const float* gru_Wih   = (const float*)d_in[12];
  const float* gru_Whh   = (const float*)d_in[13];
  const float* gru_bih   = (const float*)d_in[14];
  const float* gru_bhh   = (const float*)d_in[15];
  const float* lstm_Wih  = (const float*)d_in[16];
  const float* lstm_Whh  = (const float*)d_in[17];
  const float* lstm_bih  = (const float*)d_in[18];
  const float* lstm_bhh  = (const float*)d_in[19];
  const float* lin1_W    = (const float*)d_in[20];
  const float* lin1_b    = (const float*)d_in[21];
  const float* fc_W      = (const float*)d_in[22];
  const float* fc_b      = (const float*)d_in[23];
  const float* lin2_W    = (const float*)d_in[24];
  const float* lin2_b    = (const float*)d_in[25];

  const int* src = eidx;
  const int* dst = eidx + E;

  char* wp = (char*)d_ws;
  auto alloc = [&](size_t bytes) -> void* {
    void* p = (void*)wp;
    wp += (bytes + 255) & ~(size_t)255;
    return p;
  };

  unsigned short* xb      = (unsigned short*)alloc((size_t)N * 96 * 2);
  unsigned short* lin0Wb  = (unsigned short*)alloc((size_t)96 * 64 * 2);
  unsigned short* eb      = (unsigned short*)alloc((size_t)E * 64 * 2);
  unsigned short* enn1Wb  = (unsigned short*)alloc((size_t)3 * 64 * 128 * 2);
  unsigned short* rootWb  = (unsigned short*)alloc((size_t)3 * 64 * 64 * 2);
  unsigned short* WihB    = (unsigned short*)alloc((size_t)3 * 64 * 192 * 2);
  unsigned short* WhhB    = (unsigned short*)alloc((size_t)3 * 64 * 192 * 2);
  unsigned short* w2r     = (unsigned short*)alloc((size_t)3 * 8256 * 64 * 2);
  float*          h       = (float*)alloc((size_t)N * 64 * 4);
  unsigned short* hb      = (unsigned short*)alloc((size_t)N * 64 * 2);
  unsigned short* hiddenb = (unsigned short*)alloc((size_t)E * 128 * 2);
  unsigned short* xsrcb   = (unsigned short*)alloc((size_t)E * 64 * 2);
  float*          agg     = (float*)alloc((size_t)N * 64 * 4);
  float*          troot   = (float*)alloc((size_t)N * 64 * 4);
  unsigned short* mb      = (unsigned short*)alloc((size_t)N * 64 * 2);
  float*          gi      = (float*)alloc((size_t)N * 192 * 4);
  float*          gh      = (float*)alloc((size_t)N * 192 * 4);
  float*          deg     = (float*)alloc((size_t)N * 4);
  float*          e_buf   = (float*)alloc((size_t)N * 4);
  unsigned*       emax    = (unsigned*)alloc((size_t)Bg * 4);
  float*          denom   = (float*)alloc((size_t)Bg * 4);
  float*          rvec    = (float*)alloc((size_t)Bg * 64 * 4);
  float*          hh      = (float*)alloc((size_t)Bg * 64 * 4);
  float*          cc      = (float*)alloc((size_t)Bg * 64 * 4);
  float*          qstar   = (float*)alloc((size_t)Bg * 128 * 4);
  float*          gates   = (float*)alloc((size_t)Bg * 256 * 4);
  float*          obuf0   = (float*)alloc((size_t)Bg * 128 * 4);
  float*          obuf1   = (float*)alloc((size_t)Bg * 128 * 4);

  auto nb = [](long n, int t) { return (unsigned)((n + t - 1) / t); };
  const int T = 256;

  // ---- weight / input conversion to padded bf16 --------------------------
  k_cvt_pad<<<nb((long)N * 96, T), T, 0, stream>>>(x, xb, N, 92, N, 96);
  k_cvt_pad<<<nb(96 * 64, T), T, 0, stream>>>(lin0_W, lin0Wb, 92, 64, 96, 64);
  k_cvt_pad<<<nb((long)E * 64, T), T, 0, stream>>>(edge_attr, eb, E, 50, E, 64);
  for (int l = 0; l < 3; ++l) {
    k_cvt_pad<<<nb(64 * 128, T), T, 0, stream>>>(enn1_W + (size_t)l * 50 * 128,
                                                 enn1Wb + (size_t)l * 64 * 128, 50, 128, 64, 128);
    k_cvt_pad<<<nb(64 * 64, T), T, 0, stream>>>(root_W + (size_t)l * 64 * 64,
                                                rootWb + (size_t)l * 64 * 64, 64, 64, 64, 64);
    k_cvt_pad<<<nb(64 * 192, T), T, 0, stream>>>(gru_Wih + (size_t)l * 64 * 192,
                                                 WihB + (size_t)l * 64 * 192, 64, 192, 64, 192);
    k_cvt_pad<<<nb(64 * 192, T), T, 0, stream>>>(gru_Whh + (size_t)l * 64 * 192,
                                                 WhhB + (size_t)l * 64 * 192, 64, 192, 64, 192);
  }
  k_build_w2r<<<nb(3L * 8256 * 64, T), T, 0, stream>>>(enn2_W, enn2_b, w2r);

  // ---- lin0: h = relu(x @ lin0_W + b) ------------------------------------
  const int mtN = (N + 15) / 16, gN = (mtN + 3) / 4;     // 938 tiles, 235 blocks
  const int mtE = E / 16,        gE = (mtE + 3) / 4;     // 3750 tiles, 938 blocks
  gemm_bf16_wmma<4><<<gN, 128, 0, stream>>>(xb, lin0Wb, lin0_b, h, hb, N, 96, 96, 1);

  // ---- degree ------------------------------------------------------------
  k_zero<<<nb(N, T), T, 0, stream>>>(deg, N);
  k_deg<<<nb(E, T), T, 0, stream>>>(dst, deg, E);

  // ---- 3 message-passing layers ------------------------------------------
  for (int l = 0; l < 3; ++l) {
    gemm_bf16_wmma<8><<<gE, 128, 0, stream>>>(eb, enn1Wb + (size_t)l * 64 * 128,
                                              enn1_b + (size_t)l * 128,
                                              (float*)nullptr, hiddenb, E, 64, 64, 1);
    k_gather<<<nb((long)E * 64, T), T, 0, stream>>>(hb, src, xsrcb, E * 64);
    k_zero<<<nb((long)N * 64, T), T, 0, stream>>>(agg, N * 64);
    k_msg<<<gE, 128, 0, stream>>>(hiddenb, xsrcb, w2r + (size_t)l * 8256 * 64, dst, agg, E);
    gemm_bf16_wmma<4><<<gN, 128, 0, stream>>>(hb, rootWb + (size_t)l * 64 * 64,
                                              (float*)nullptr, troot,
                                              (unsigned short*)nullptr, N, 64, 64, 0);
    k_m<<<nb((long)N * 64, T), T, 0, stream>>>(agg, deg, troot, conv_b + (size_t)l * 64,
                                               mb, N * 64);
    gemm_bf16_wmma<12><<<gN, 128, 0, stream>>>(mb, WihB + (size_t)l * 64 * 192,
                                               gru_bih + (size_t)l * 192, gi,
                                               (unsigned short*)nullptr, N, 64, 64, 0);
    gemm_bf16_wmma<12><<<gN, 128, 0, stream>>>(hb, WhhB + (size_t)l * 64 * 192,
                                               gru_bhh + (size_t)l * 192, gh,
                                               (unsigned short*)nullptr, N, 64, 64, 0);
    k_gru<<<nb((long)N * 64, T), T, 0, stream>>>(gi, gh, h, hb, N * 64);
  }

  // ---- Set2Set pooling (3 steps) -----------------------------------------
  k_zero<<<nb(Bg * 64, T), T, 0, stream>>>(hh, Bg * 64);
  k_zero<<<nb(Bg * 64, T), T, 0, stream>>>(cc, Bg * 64);
  k_zero<<<nb(Bg * 128, T), T, 0, stream>>>(qstar, Bg * 128);
  for (int it = 0; it < 3; ++it) {
    k_dense<<<nb(Bg * 256, T), T, 0, stream>>>(qstar, lstm_Wih, lstm_bih, gates,
                                               Bg, 128, 256, 0, 0);
    k_dense<<<nb(Bg * 256, T), T, 0, stream>>>(hh, lstm_Whh, lstm_bhh, gates,
                                               Bg, 64, 256, 0, 1);
    k_lstm_update<<<nb(Bg * 64, T), T, 0, stream>>>(gates, hh, cc, Bg * 64);
    k_zero<<<1, T, 0, stream>>>((float*)emax, Bg);   // enc(0)=0 acts as -inf sentinel
    k_zero<<<1, T, 0, stream>>>(denom, Bg);
    k_zero<<<nb(Bg * 64, T), T, 0, stream>>>(rvec, Bg * 64);
    k_attn1<<<nb(N, T), T, 0, stream>>>(h, hh, batch, e_buf, emax, N);
    k_attn2<<<nb(N, T), T, 0, stream>>>(e_buf, emax, batch, denom, N);
    k_attn3<<<nb((long)N * 64, T), T, 0, stream>>>(e_buf, denom, batch, h, rvec, N * 64);
    k_qstar<<<nb(Bg * 128, T), T, 0, stream>>>(hh, rvec, qstar, Bg * 128);
  }

  // ---- head: lin1 -> 2x fc -> lin2 ---------------------------------------
  k_dense<<<nb(Bg * 128, T), T, 0, stream>>>(qstar, lin1_W, lin1_b, obuf0, Bg, 128, 128, 1, 0);
  k_dense<<<nb(Bg * 128, T), T, 0, stream>>>(obuf0, fc_W, fc_b, obuf1, Bg, 128, 128, 1, 0);
  k_dense<<<nb(Bg * 128, T), T, 0, stream>>>(obuf1, fc_W + 128 * 128, fc_b + 128, obuf0,
                                             Bg, 128, 128, 1, 0);
  k_dense<<<nb(Bg, T), T, 0, stream>>>(obuf0, lin2_W, lin2_b, (float*)d_out, Bg, 128, 1, 0, 0);
}
